// GATFactVerifier_13649406067342
// MI455X (gfx1250) — matrix operand order repaired and checked by reading, not compiled, not run on hardware
//
#include <hip/hip_runtime.h>
#include <hip/hip_bf16.h>

typedef __attribute__((ext_vector_type(16))) __bf16 v16bf;
typedef __attribute__((ext_vector_type(8)))  float  v8f;

#define Nn   50000
#define Ee   1600000
#define ETOT 1650000      // edges + self loops
#define Hh   4
#define Cc   32
#define HC   128
#define Bb   512
#define NCLS 4
#define NEG  0.2f

union FragB { v16bf v; uint4 q[2]; };

// monotonic float <-> uint key for atomicMax-based segment max
__device__ __forceinline__ unsigned fkey(float f) {
  unsigned b = __float_as_uint(f);
  return (b & 0x80000000u) ? ~b : (b | 0x80000000u);
}
__device__ __forceinline__ float funkey(unsigned k) {
  unsigned b = (k & 0x80000000u) ? (k & 0x7FFFFFFFu) : ~k;
  return __uint_as_float(b);
}

// ---- weights: fp32 [K,N] -> bf16 transposed [N,K] (both layers) ----
__global__ void k_prep_w(const float* __restrict__ w1, const float* __restrict__ w2,
                         __bf16* __restrict__ w1t, __bf16* __restrict__ w2t) {
  int t = blockIdx.x * blockDim.x + threadIdx.x;
  if (t >= HC * HC) return;
  int k = t / HC, n = t % HC;
  w1t[n * HC + k] = (__bf16)w1[k * HC + n];
  w2t[n * HC + k] = (__bf16)w2[k * HC + n];
}

__global__ void k_cast_bf16(const float* __restrict__ x, __bf16* __restrict__ xb, int cnt) {
  int t = blockIdx.x * blockDim.x + threadIdx.x;
  if (t < cnt) xb[t] = (__bf16)x[t];
}

__global__ void k_zero_f32(float* __restrict__ p, int cnt) {
  int t = blockIdx.x * blockDim.x + threadIdx.x;
  if (t < cnt) p[t] = 0.0f;
}

__global__ void k_init_nh(unsigned* __restrict__ menc, float* __restrict__ den) {
  int t = blockIdx.x * blockDim.x + threadIdx.x;
  if (t >= Nn * Hh) return;
  menc[t] = 0u;        // fkey(-inf) > 0 would also work; 0 is below every real key
  den[t]  = 0.0f;
}

// ---- WMMA GEMM: Out[nrows,128] = A[nrows,128](bf16) @ W(128,128) via WT[n][k] in LDS ----
__global__ __launch_bounds__(256)
void k_gemm(const __bf16* __restrict__ A, const __bf16* __restrict__ WT,
            float* __restrict__ Out, int nrows) {
  __shared__ __bf16 ldsw[HC * HC];   // 32 KB, WT layout [n][k]
  const int tid = threadIdx.x;
  {
    const uint4* s = (const uint4*)WT;
    uint4* dp = (uint4*)ldsw;
#pragma unroll
    for (int i = 0; i < 8; ++i) dp[tid + i * 256] = s[tid + i * 256];
  }
  __syncthreads();

  const int lane  = tid & 31;
  const int wave  = tid >> 5;
  const int strip = blockIdx.x * 8 + wave;      // 16-row output strip per wave
  if (strip * 16 >= nrows) return;

  const int m    = lane & 15;
  const int half = lane >> 4;
  const __bf16* arow = A + (size_t)(strip * 16 + m) * HC;

  v8f acc[8];
#pragma unroll
  for (int i = 0; i < 8; ++i) {
#pragma unroll
    for (int j = 0; j < 8; ++j) acc[i][j] = 0.0f;
  }

#pragma unroll
  for (int kk = 0; kk < 4; ++kk) {
    // A frag (ISA 16-bit A 16x32): lanes 0-15 K{0..7,16..23}, lanes 16-31 K{8..15,24..31}
    FragB a;
    const int k0 = kk * 32 + half * 8;
    a.q[0] = *(const uint4*)(arow + k0);
    a.q[1] = *(const uint4*)(arow + k0 + 16);
#pragma unroll
    for (int nt = 0; nt < 8; ++nt) {
      // B frag: lane<16 -> col N=lane, K{0..7,16..23} ; lane>=16 -> col N=lane-16, K{8..15,24..31}
      FragB b;
      const __bf16* bp = ldsw + (size_t)(nt * 16 + m) * HC + kk * 32 + half * 8;
      b.q[0] = *(const uint4*)(bp);
      b.q[1] = *(const uint4*)(bp + 16);
      acc[nt] = __builtin_amdgcn_wmma_f32_16x16x32_bf16(
          false, a.v, false, b.v, (short)0, acc[nt], false, false);
    }
  }

  // C/D layout: VGPR r -> row (half*8 + r), col = nt*16 + (lane&15)
  float* obase = Out + (size_t)strip * 16 * HC;
#pragma unroll
  for (int nt = 0; nt < 8; ++nt) {
    const int col = nt * 16 + m;
#pragma unroll
    for (int r = 0; r < 8; ++r)
      obase[(half * 8 + r) * HC + col] = acc[nt][r];
  }
}

// ---- per-(node,head) attention score dots ----
__global__ void k_scores(const float* __restrict__ h, const float* __restrict__ as,
                         const float* __restrict__ ad, float* __restrict__ es,
                         float* __restrict__ ed) {
  int t = blockIdx.x * blockDim.x + threadIdx.x;
  if (t >= Nn * Hh) return;
  int n = t >> 2, hh = t & 3;
  const float* row = h + (size_t)n * HC + hh * Cc;
  const float* a1 = as + hh * Cc;
  const float* a2 = ad + hh * Cc;
  float s = 0.f, d = 0.f;
#pragma unroll
  for (int c = 0; c < Cc; ++c) { s += row[c] * a1[c]; d += row[c] * a2[c]; }
  es[t] = s; ed[t] = d;
}

__device__ __forceinline__ void edge_sd(const int* ei, int e, int& s, int& d) {
  if (e < Ee) { s = ei[e]; d = ei[Ee + e]; }
  else        { s = e - Ee; d = e - Ee; }        // self loops
}

__global__ void k_edge_max(const int* __restrict__ ei, const float* __restrict__ es,
                           const float* __restrict__ ed, unsigned* __restrict__ menc) {
  int t = blockIdx.x * blockDim.x + threadIdx.x;
  if (t >= ETOT * Hh) return;
  int e = t >> 2, hh = t & 3, s, d;
  edge_sd(ei, e, s, d);
  float v = es[s * Hh + hh] + ed[d * Hh + hh];
  v = v > 0.f ? v : NEG * v;
  atomicMax(&menc[d * Hh + hh], fkey(v));
}

__global__ void k_edge_den(const int* __restrict__ ei, const float* __restrict__ es,
                           const float* __restrict__ ed, const unsigned* __restrict__ menc,
                           float* __restrict__ den) {
  int t = blockIdx.x * blockDim.x + threadIdx.x;
  if (t >= ETOT * Hh) return;
  int e = t >> 2, hh = t & 3, s, d;
  edge_sd(ei, e, s, d);
  int nh = d * Hh + hh;
  float v = es[s * Hh + hh] + ed[nh];
  v = v > 0.f ? v : NEG * v;
  atomicAdd(&den[nh], __expf(v - funkey(menc[nh])));
}

__global__ void k_edge_agg(const int* __restrict__ ei, const float* __restrict__ es,
                           const float* __restrict__ ed, const unsigned* __restrict__ menc,
                           const float* __restrict__ den, const float* __restrict__ feat,
                           float* __restrict__ agg) {
  int t = blockIdx.x * blockDim.x + threadIdx.x;     // ETOT*128 = 211.2M < 2^31
  if (t >= ETOT * HC) return;
  int e = t >> 7, c = t & 127, hh = c >> 5, s, d;
  edge_sd(ei, e, s, d);
  int nh = d * Hh + hh;
  float v = es[s * Hh + hh] + ed[nh];
  v = v > 0.f ? v : NEG * v;
  float alpha = __expf(v - funkey(menc[nh])) / (den[nh] + 1e-16f);
  atomicAdd(&agg[(size_t)d * HC + c], alpha * feat[(size_t)s * HC + c]);
}

__global__ void k_bias_act(float* __restrict__ agg, const float* __restrict__ bias,
                           __bf16* __restrict__ xb, int dorelu, int docast) {
  int t = blockIdx.x * blockDim.x + threadIdx.x;
  if (t >= Nn * HC) return;
  float v = agg[t] + bias[t & 127];
  if (dorelu) v = fmaxf(v, 0.f);
  agg[t] = v;
  if (docast) xb[t] = (__bf16)v;
}

__global__ void k_pool(const float* __restrict__ agg, const int* __restrict__ batch,
                       float* __restrict__ sums, float* __restrict__ cnt) {
  int t = blockIdx.x * blockDim.x + threadIdx.x;
  if (t >= Nn * HC) return;
  int n = t >> 7, c = t & 127;
  int b = batch[n];
  atomicAdd(&sums[b * HC + c], agg[t]);
  if (c == 0) atomicAdd(&cnt[b], 1.0f);
}

__global__ void k_classify(const float* __restrict__ sums, const float* __restrict__ cnt,
                           const float* __restrict__ wc, const float* __restrict__ bc,
                           float* __restrict__ out) {
  int t = blockIdx.x * blockDim.x + threadIdx.x;
  if (t >= Bb * NCLS) return;
  int b = t >> 2, o = t & 3;
  float inv = 1.0f / fmaxf(cnt[b], 1.0f);
  float acc = bc[o];
#pragma unroll 8
  for (int k = 0; k < HC; ++k) acc += sums[b * HC + k] * inv * wc[k * NCLS + o];
  out[t] = acc;
}

extern "C" void kernel_launch(void* const* d_in, const int* in_sizes, int n_in,
                              void* d_out, int out_size, void* d_ws, size_t ws_size,
                              hipStream_t stream) {
  const float* x   = (const float*)d_in[0];
  const int*   ei  = (const int*)  d_in[1];
  const int*   bat = (const int*)  d_in[2];
  const float* w1  = (const float*)d_in[3];
  const float* as1 = (const float*)d_in[4];
  const float* ad1 = (const float*)d_in[5];
  const float* b1  = (const float*)d_in[6];
  const float* w2  = (const float*)d_in[7];
  const float* as2 = (const float*)d_in[8];
  const float* ad2 = (const float*)d_in[9];
  const float* b2  = (const float*)d_in[10];
  const float* wc  = (const float*)d_in[11];
  const float* bc  = (const float*)d_in[12];
  float* out = (float*)d_out;

  char* ws = (char*)d_ws;
  size_t off = 0;
  auto take = [&](size_t bytes) -> char* {
    char* p = ws + off;
    off += (bytes + 255) & ~(size_t)255;
    return p;
  };
  __bf16*   xb   = (__bf16*)  take((size_t)Nn * HC * 2);
  __bf16*   w1t  = (__bf16*)  take((size_t)HC * HC * 2);
  __bf16*   w2t  = (__bf16*)  take((size_t)HC * HC * 2);
  float*    hbuf = (float*)   take((size_t)Nn * HC * 4);
  float*    agg  = (float*)   take((size_t)Nn * HC * 4);
  float*    es   = (float*)   take((size_t)Nn * Hh * 4);
  float*    ed   = (float*)   take((size_t)Nn * Hh * 4);
  float*    den  = (float*)   take((size_t)Nn * Hh * 4);
  unsigned* menc = (unsigned*)take((size_t)Nn * Hh * 4);
  float*    sums = (float*)   take((size_t)Bb * HC * 4);
  float*    cnt  = (float*)   take((size_t)Bb * 4);
  (void)ws_size; (void)n_in; (void)in_sizes; (void)out_size;

  const int TB = 256;
  auto blks = [](long long n) { return (unsigned)((n + 255) / 256); };
  const unsigned gNH  = blks((long long)Nn * Hh);
  const unsigned gNC  = blks((long long)Nn * HC);
  const unsigned gEH  = blks((long long)ETOT * Hh);
  const unsigned gEC  = blks((long long)ETOT * HC);
  const unsigned gGEM = (Nn / 16 + 7) / 8;          // 3125 strips / 8 waves per block

  k_prep_w<<<blks(HC * HC), TB, 0, stream>>>(w1, w2, w1t, w2t);
  k_cast_bf16<<<gNC, TB, 0, stream>>>(x, xb, Nn * HC);

  // ---------------- layer 1 ----------------
  k_gemm<<<gGEM, TB, 0, stream>>>(xb, w1t, hbuf, Nn);
  k_scores<<<gNH, TB, 0, stream>>>(hbuf, as1, ad1, es, ed);
  k_init_nh<<<gNH, TB, 0, stream>>>(menc, den);
  k_zero_f32<<<gNC, TB, 0, stream>>>(agg, Nn * HC);
  k_edge_max<<<gEH, TB, 0, stream>>>(ei, es, ed, menc);
  k_edge_den<<<gEH, TB, 0, stream>>>(ei, es, ed, menc, den);
  k_edge_agg<<<gEC, TB, 0, stream>>>(ei, es, ed, menc, den, hbuf, agg);
  k_bias_act<<<gNC, TB, 0, stream>>>(agg, b1, xb, /*relu=*/1, /*cast=*/1);

  // ---------------- layer 2 ----------------
  k_gemm<<<gGEM, TB, 0, stream>>>(xb, w2t, hbuf, Nn);
  k_scores<<<gNH, TB, 0, stream>>>(hbuf, as2, ad2, es, ed);
  k_init_nh<<<gNH, TB, 0, stream>>>(menc, den);
  k_zero_f32<<<gNC, TB, 0, stream>>>(agg, Nn * HC);
  k_edge_max<<<gEH, TB, 0, stream>>>(ei, es, ed, menc);
  k_edge_den<<<gEH, TB, 0, stream>>>(ei, es, ed, menc, den);
  k_edge_agg<<<gEC, TB, 0, stream>>>(ei, es, ed, menc, den, hbuf, agg);
  k_bias_act<<<gNC, TB, 0, stream>>>(agg, b2, xb, /*relu=*/0, /*cast=*/0);

  // ---------------- pool + classify ----------------
  k_zero_f32<<<blks(Bb * HC), TB, 0, stream>>>(sums, Bb * HC);
  k_zero_f32<<<blks(Bb), TB, 0, stream>>>(cnt, Bb);
  k_pool<<<gNC, TB, 0, stream>>>(agg, bat, sums, cnt);
  k_classify<<<blks(Bb * NCLS), TB, 0, stream>>>(sums, cnt, wc, bc, out);
}
